// DynamicGCN_38800734552216
// MI455X (gfx1250) — matrix operand order repaired and checked by reading, not compiled, not run on hardware
//
#include <hip/hip_runtime.h>

// ---------------------------------------------------------------------------
// DynamicGCN on MI455X (gfx1250, wave32, WMMA)
//   layer: H = A_norm @ (X @ W) + b ; relu fused into next GEMM's input cvt.
//   GEMMs: v_wmma_f32_16x16x32_f16 (f32 accumulate), weights pre-transposed
//   to f16 [N][K] so B-fragments are contiguous 16B loads (hot in WGP$/L2).
//   GEMM epilogue fuses agg = H*isd^2 + b (self-loop+bias) -> saves a 51MB
//   read pass per layer. Scatter: wave-per-edge float4 gathers +
//   global_atomic_add_f32; 51.2MB feature matrix is L2-resident (192MB L2).
//   Layer-2 uses a separate agg2 buffer (no input/output aliasing across
//   concurrently-running blocks).
// ---------------------------------------------------------------------------

typedef __attribute__((ext_vector_type(16))) _Float16 v16h;
typedef __attribute__((ext_vector_type(8)))  _Float16 v8h;
typedef __attribute__((ext_vector_type(8)))  float    v8f;

#define DFEAT 256
#define HID   128
#define ODIM  64

static inline int cdiv(long long a, long long b) { return (int)((a + b - 1) / b); }

// deg[i] = 1 (self loop)
__global__ void k_init_deg(float* __restrict__ deg, int n) {
  int i = blockIdx.x * blockDim.x + threadIdx.x;
  if (i < n) deg[i] = 1.0f;
}

// deg[dst[e]] += 1
__global__ void k_deg_accum(float* __restrict__ deg, const int* __restrict__ dst, int e) {
  int i = blockIdx.x * blockDim.x + threadIdx.x;
  if (i < e) atomicAdd(&deg[dst[i]], 1.0f);
}

// in-place inv_sqrt_deg
__global__ void k_rsqrt(float* __restrict__ d, int n) {
  int i = blockIdx.x * blockDim.x + threadIdx.x;
  if (i < n) d[i] = __frsqrt_rn(d[i]);
}

// Wt[n][k] = (f16) W[k][n]   (tiny, runs once per layer)
__global__ void k_wcvt(const float* __restrict__ W, _Float16* __restrict__ Wt, int K, int N) {
  int t = blockIdx.x * blockDim.x + threadIdx.x;
  if (t >= K * N) return;
  int k = t / N, n = t - k * N;
  Wt[n * K + k] = (_Float16)W[t];
}

// wave-per-edge scatter: agg[dst] += H[src] * (isd[src]*isd[dst])
__global__ void k_edge_scatter(const float* __restrict__ H, const int* __restrict__ src,
                               const int* __restrict__ dst, const float* __restrict__ isd,
                               float* __restrict__ agg, int e) {
  long long gid = (long long)blockIdx.x * blockDim.x + threadIdx.x;
  int ed = (int)(gid >> 5);
  if (ed >= e) return;
  int q = (int)(gid & 31);
  int s = src[ed], d = dst[ed];
  float nrm = isd[s] * isd[d];
  float4 h = ((const float4*)(H + (size_t)s * HID))[q];   // coalesced 512B/edge
  float* po = agg + (size_t)d * HID + q * 4;
  atomicAdd(po + 0, h.x * nrm);
  atomicAdd(po + 1, h.y * nrm);
  atomicAdd(po + 2, h.z * nrm);
  atomicAdd(po + 3, h.w * nrm);
}

// C[M x N] = op(A[M x K]) @ Bt^T  (+ bias), Bt is f16 [N][K] (col-major W).
// FUSE: additionally write agg = C * isd[row]^2 + bias[col] (self-loop+bias).
// Block = 128 threads = 4 waves; each wave owns a 16 x N strip.
template<int K, int N, bool RELU_IN, bool BIAS, bool FUSE>
__launch_bounds__(128)
__global__ void k_gemm_wmma(const float* __restrict__ A, const _Float16* __restrict__ Bt,
                            const float* __restrict__ bias, float* __restrict__ C,
                            float* __restrict__ agg, const float* __restrict__ isd, int M) {
  constexpr int NT = N / 16;
  const int lane = threadIdx.x & 31;
  const int wave = threadIdx.x >> 5;
  const int row0 = (blockIdx.x * 4 + wave) * 16;
  if (row0 >= M) return;                 // wave-uniform: EXEC stays all-1s for WMMA

  const int m  = lane & 15;
  const int kh = lane >> 4;
  int rowA = row0 + m; if (rowA >= M) rowA = M - 1;   // clamp for partial tile
  const float* pa = A + (size_t)rowA * K;

  v8f zero = {};
  v8f acc[NT];
#pragma unroll
  for (int i = 0; i < NT; ++i) acc[i] = zero;

  const int nb = lane & 15;

  for (int kb = 0; kb < K; kb += 32) {
    // --- A fragment: a[j] = A[m][kb + (j<8?0:16) + kh*8 + (j&7)] ---
    float4 f0 = *(const float4*)(pa + kb + kh * 8);
    float4 f1 = *(const float4*)(pa + kb + kh * 8 + 4);
    float4 f2 = *(const float4*)(pa + kb + 16 + kh * 8);
    float4 f3 = *(const float4*)(pa + kb + 16 + kh * 8 + 4);
    if constexpr (RELU_IN) {
      f0.x = fmaxf(f0.x, 0.f); f0.y = fmaxf(f0.y, 0.f); f0.z = fmaxf(f0.z, 0.f); f0.w = fmaxf(f0.w, 0.f);
      f1.x = fmaxf(f1.x, 0.f); f1.y = fmaxf(f1.y, 0.f); f1.z = fmaxf(f1.z, 0.f); f1.w = fmaxf(f1.w, 0.f);
      f2.x = fmaxf(f2.x, 0.f); f2.y = fmaxf(f2.y, 0.f); f2.z = fmaxf(f2.z, 0.f); f2.w = fmaxf(f2.w, 0.f);
      f3.x = fmaxf(f3.x, 0.f); f3.y = fmaxf(f3.y, 0.f); f3.z = fmaxf(f3.z, 0.f); f3.w = fmaxf(f3.w, 0.f);
    }
    v16h a;
    a[0]  = (_Float16)f0.x; a[1]  = (_Float16)f0.y; a[2]  = (_Float16)f0.z; a[3]  = (_Float16)f0.w;
    a[4]  = (_Float16)f1.x; a[5]  = (_Float16)f1.y; a[6]  = (_Float16)f1.z; a[7]  = (_Float16)f1.w;
    a[8]  = (_Float16)f2.x; a[9]  = (_Float16)f2.y; a[10] = (_Float16)f2.z; a[11] = (_Float16)f2.w;
    a[12] = (_Float16)f3.x; a[13] = (_Float16)f3.y; a[14] = (_Float16)f3.z; a[15] = (_Float16)f3.w;

    // --- preload all B fragments for this K-step (clause the loads) ---
    // b[j] = Bt[nt*16+nb][kb + (j<8?0:16) + kh*8 + (j&7)], two 16B loads per fragment
    v16h bfrag[NT];
#pragma unroll
    for (int nt = 0; nt < NT; ++nt) {
      const _Float16* pb = Bt + (size_t)(nt * 16 + nb) * K + kb + kh * 8;
      v8h b0 = *(const v8h*)(pb);
      v8h b1 = *(const v8h*)(pb + 16);
#pragma unroll
      for (int j = 0; j < 8; ++j) { bfrag[nt][j] = b0[j]; bfrag[nt][8 + j] = b1[j]; }
    }

#pragma unroll
    for (int nt = 0; nt < NT; ++nt) {
      acc[nt] = __builtin_amdgcn_wmma_f32_16x16x32_f16(
          false, a, false, bfrag[nt], (short)0, acc[nt], false, false);
    }
  }

  // --- epilogue: acc[nt][r] -> row = row0 + (lane<16? r : 8+r), col = nt*16+nb ---
  const int rbase = row0 + (lane < 16 ? 0 : 8);

  float wsq[8];
  if constexpr (FUSE) {
#pragma unroll
    for (int r = 0; r < 8; ++r) {
      int row = rbase + r; if (row >= M) row = M - 1;
      float w = isd[row];
      wsq[r] = w * w;
    }
  }

  if (row0 + 16 <= M) {
    // fast path: full tile, no per-element guards (1562 of 1563 blocks)
#pragma unroll
    for (int nt = 0; nt < NT; ++nt) {
      const int col = nt * 16 + nb;
      const float bv = (BIAS || FUSE) ? bias[col] : 0.0f;
#pragma unroll
      for (int r = 0; r < 8; ++r) {
        const size_t idx = (size_t)(rbase + r) * N + col;
        const float v = acc[nt][r];
        if constexpr (BIAS) C[idx] = v + bv;
        else                C[idx] = v;
        if constexpr (FUSE) agg[idx] = v * wsq[r] + bv;
      }
    }
  } else {
#pragma unroll
    for (int nt = 0; nt < NT; ++nt) {
      const int col = nt * 16 + nb;
      const float bv = (BIAS || FUSE) ? bias[col] : 0.0f;
#pragma unroll
      for (int r = 0; r < 8; ++r) {
        const int row = rbase + r;
        if (row < M) {
          const size_t idx = (size_t)row * N + col;
          const float v = acc[nt][r];
          if constexpr (BIAS) C[idx] = v + bv;
          else                C[idx] = v;
          if constexpr (FUSE) agg[idx] = v * wsq[r] + bv;
        }
      }
    }
  }
}

extern "C" void kernel_launch(void* const* d_in, const int* in_sizes, int n_in,
                              void* d_out, int out_size, void* d_ws, size_t ws_size,
                              hipStream_t stream) {
  const float* x   = (const float*)d_in[0];
  const int*   ei  = (const int*)  d_in[1];   // [2, E] int32 (harness convention)
  const float* W1  = (const float*)d_in[3];
  const float* b1  = (const float*)d_in[4];
  const float* W2  = (const float*)d_in[5];
  const float* b2  = (const float*)d_in[6];
  const float* Wfc = (const float*)d_in[7];
  const float* bfc = (const float*)d_in[8];
  float* out = (float*)d_out;

  const int Nn = in_sizes[0] / DFEAT;         // 100000
  const int E  = in_sizes[1] / 2;             // 1600000
  const int* e_src = ei;
  const int* e_dst = ei + E;

  // workspace carve (d_ws is 256B-aligned)
  char* w = (char*)d_ws;
  float* isd  = (float*)w;  w += (((size_t)Nn * 4 + 255) & ~(size_t)255);
  float* H    = (float*)w;  w += (size_t)Nn * HID * 4;          // 51.2 MB
  float* agg  = (float*)w;  w += (size_t)Nn * HID * 4;          // 51.2 MB
  float* agg2 = (float*)w;  w += (size_t)Nn * HID * 4;          // 51.2 MB (layer-2 out)
  _Float16* Wt1 = (_Float16*)w; w += (size_t)DFEAT * HID * 2;
  _Float16* Wt2 = (_Float16*)w; w += (size_t)HID * HID * 2;
  _Float16* Wtf = (_Float16*)w; w += (size_t)HID * ODIM * 2;

  const int TB = 256;

  // degree / normalization
  k_init_deg <<<cdiv(Nn, TB), TB, 0, stream>>>(isd, Nn);
  k_deg_accum<<<cdiv(E,  TB), TB, 0, stream>>>(isd, e_dst, E);
  k_rsqrt    <<<cdiv(Nn, TB), TB, 0, stream>>>(isd, Nn);

  // weight transpose+cvt to f16 [N][K]
  k_wcvt<<<cdiv(DFEAT * HID, TB), TB, 0, stream>>>(W1,  Wt1, DFEAT, HID);
  k_wcvt<<<cdiv(HID * HID,   TB), TB, 0, stream>>>(W2,  Wt2, HID,   HID);
  k_wcvt<<<cdiv(HID * ODIM,  TB), TB, 0, stream>>>(Wfc, Wtf, HID,   ODIM);

  const int gemm_blocks = cdiv(Nn, 64);       // 4 waves x 16 rows per block

  // layer 1: H = X @ W1 (epilogue: agg = H*isd^2 + b1) ; agg += A_norm-scatter
  k_gemm_wmma<DFEAT, HID, false, false, true><<<gemm_blocks, 128, 0, stream>>>(
      x, Wt1, b1, H, agg, isd, Nn);
  k_edge_scatter<<<cdiv((long long)E * 32, TB), TB, 0, stream>>>(H, e_src, e_dst, isd, agg, E);

  // layer 2: H = relu(agg) @ W2 (epilogue: agg2 = H*isd^2 + b2) ; scatter into agg2
  k_gemm_wmma<HID, HID, true, false, true><<<gemm_blocks, 128, 0, stream>>>(
      agg, Wt2, b2, H, agg2, isd, Nn);
  k_edge_scatter<<<cdiv((long long)E * 32, TB), TB, 0, stream>>>(H, e_src, e_dst, isd, agg2, E);

  // head: out = relu(agg2) @ Wfc + bfc
  k_gemm_wmma<HID, ODIM, true, true, false><<<gemm_blocks, 128, 0, stream>>>(
      agg2, Wtf, bfc, out, nullptr, nullptr, Nn);
}
